// Model_19172734009568
// MI455X (gfx1250) — matrix operand order repaired and checked by reading, not compiled, not run on hardware
//
#include <hip/hip_runtime.h>
#include <hip/hip_fp16.h>

// ---------------------------------------------------------------------------
// KAN layer for MI455X (gfx1250, wave32).
//   out[b,o] = sum_d silu(x[b,d]) * Wb[d,o]  +  sum_d P_{d,s(x),o}(x)
// where P is the per-(d, knot-span, o) cubic polynomial obtained by folding
// coef*scale_sp*mask into the Cox-de Boor basis restricted to that span.
//
// Kernels:
//   kan_prep       : symbolic Cox-de Boor -> cubic coeff table (VALU, tiny)
//   kan_spline     : hot path; span search + Horner, partials to ws
//   kan_base_wmma  : out_base GEMM via v_wmma_f32_16x16x32_f16
//   kan_reduce     : deterministic sum of 33 partials -> d_out
// ---------------------------------------------------------------------------

typedef __attribute__((ext_vector_type(16))) _Float16 v16h;
typedef __attribute__((ext_vector_type(8)))  float    v8f;

#define BATCH 2048
#define DIN   4096
#define OOUT  3
#define GKNOT 12
#define NSPL  8      // NUM + K basis functions per d
#define NSEG  6      // knot spans s-3 in [0,5]

// workspace layout (float offsets)
#define APOLY_F 0              // [DIN][NSEG][OOUT][4]  = 294912 floats
#define KNOTS_F 294912         // [DIN][8]              =  32768 floats
#define WBP_F   327680         // [DIN][4]              =  16384 floats
#define PART_F  344064         // [33][BATCH][OOUT]     = 202752 floats
#define NPART   33

// ------------------------------ prep ---------------------------------------
__global__ __launch_bounds__(256) void kan_prep(
    const float* __restrict__ grid, const float* __restrict__ coef,
    const float* __restrict__ scale_base, const float* __restrict__ scale_sp,
    const float* __restrict__ mask, float* __restrict__ ws)
{
  int tid = blockIdx.x * blockDim.x + threadIdx.x;
  if (tid >= DIN * NSEG) return;
  int d  = tid / NSEG;
  int si = tid - d * NSEG;
  int s  = si + 3;                       // active span: [t_s, t_{s+1})

  float t[GKNOT];
#pragma unroll
  for (int i = 0; i < GKNOT; ++i) t[i] = grid[d * GKNOT + i];

  // cur[i][c]: polynomial coeffs (c0..c3) of B^p_{s-p+i} restricted to span s
  float cur[4][4];
#pragma unroll
  for (int i = 0; i < 4; ++i)
#pragma unroll
    for (int c = 0; c < 4; ++c) cur[i][c] = 0.0f;
  cur[0][0] = 1.0f;                      // level 0 indicator on span s

#pragma unroll
  for (int p = 1; p <= 3; ++p) {
    float nxt[4][4];
#pragma unroll
    for (int i = 0; i < 4; ++i)
#pragma unroll
      for (int c = 0; c < 4; ++c) nxt[i][c] = 0.0f;
    for (int i = 0; i <= p; ++i) {
      int j = s - p + i;
      if (i >= 1) {                      // ((x - t_j)/(t_{j+p}-t_j)) * B^{p-1}_j
        float inv = 1.0f / (t[j + p] - t[j]);
        float a = -t[j] * inv, bb = inv;
        nxt[i][0] += a * cur[i-1][0];
        nxt[i][1] += a * cur[i-1][1] + bb * cur[i-1][0];
        nxt[i][2] += a * cur[i-1][2] + bb * cur[i-1][1];
        nxt[i][3] += a * cur[i-1][3] + bb * cur[i-1][2];
      }
      if (i <= p - 1) {                  // ((t_{j+p+1}-x)/(t_{j+p+1}-t_{j+1})) * B^{p-1}_{j+1}
        float inv = 1.0f / (t[j + p + 1] - t[j + 1]);
        float a = t[j + p + 1] * inv, bb = -inv;
        nxt[i][0] += a * cur[i][0];
        nxt[i][1] += a * cur[i][1] + bb * cur[i][0];
        nxt[i][2] += a * cur[i][2] + bb * cur[i][1];
        nxt[i][3] += a * cur[i][3] + bb * cur[i][2];
      }
    }
#pragma unroll
    for (int i = 0; i < 4; ++i)
#pragma unroll
      for (int c = 0; c < 4; ++c) cur[i][c] = nxt[i][c];
  }

  // fold coef * scale_sp * mask into one cubic per output
#pragma unroll
  for (int o = 0; o < OOUT; ++o) {
    float sm = scale_sp[d * OOUT + o] * mask[d * OOUT + o];
    float A0 = 0.f, A1 = 0.f, A2 = 0.f, A3 = 0.f;
#pragma unroll
    for (int i = 0; i < 4; ++i) {
      int j = s - 3 + i;
      float wv = (j < NSPL) ? coef[(d * OOUT + o) * NSPL + j] * sm : 0.0f;
      A0 += cur[i][0] * wv; A1 += cur[i][1] * wv;
      A2 += cur[i][2] * wv; A3 += cur[i][3] * wv;
    }
    float4 st = make_float4(A0, A1, A2, A3);
    *(float4*)(ws + APOLY_F + ((size_t)(d * NSEG + si) * OOUT + o) * 4) = st;
  }

  if (si == 0) {
    float* kp = ws + KNOTS_F + (size_t)d * 8;   // interior knots t4..t8 for span search
    kp[0] = t[4]; kp[1] = t[5]; kp[2] = t[6]; kp[3] = t[7];
    kp[4] = t[8]; kp[5] = 0.f; kp[6] = 0.f; kp[7] = 0.f;
    float* wb = ws + WBP_F + (size_t)d * 4;     // scale_base*mask, padded
    wb[0] = scale_base[d * 3 + 0] * mask[d * 3 + 0];
    wb[1] = scale_base[d * 3 + 1] * mask[d * 3 + 1];
    wb[2] = scale_base[d * 3 + 2] * mask[d * 3 + 2];
    wb[3] = 0.f;
  }
}

// --------------------------- spline hot path --------------------------------
// block = 256 threads (8 waves). Each block: one 128-wide d-chunk x 256 b rows.
// grid = 8 b-groups * 32 d-chunks = 256 blocks. x is read exactly once.
// Knot vectors for each lane's 4 fixed d-columns are hoisted out of the b-loop;
// the inner loop is 1 b128 x-load + 12 L0-resident b128 gathers + Horner FMAs.
__global__ __launch_bounds__(256) void kan_spline(
    const float* __restrict__ x, const float* __restrict__ tab,
    float* __restrict__ part)
{
  int bx = blockIdx.x;
  int dchunk = bx & 31;
  int bgroup = bx >> 5;
  int dbase  = dchunk << 7;
  int tid  = threadIdx.x;
  int lane = tid & 31;
  int w    = tid >> 5;

  const float* apoly = tab + APOLY_F;
  const float* knots = tab + KNOTS_F;

  // warm this block's table chunk (36.9KB poly + 4KB knots) -> global_prefetch_b8
  __builtin_prefetch((const char*)(apoly + (size_t)dbase * 72) + tid * 144, 0, 1);
  if (tid < 32)
    __builtin_prefetch((const char*)(knots + (size_t)dbase * 8) + tid * 128, 0, 1);

  int d0 = dbase + lane * 4;             // this lane's 4 consecutive d's

  // hoist the per-d knot vectors (invariant across the b loop)
  float4 kva[4];
  float  k8a[4];
#pragma unroll
  for (int j = 0; j < 4; ++j) {
    kva[j] = *(const float4*)(knots + (size_t)(d0 + j) * 8);
    k8a[j] = knots[(size_t)(d0 + j) * 8 + 4];
  }

  for (int ib = 0; ib < 32; ++ib) {
    int b = (bgroup << 8) + (ib << 3) + w;
    float4 x4 = *(const float4*)(x + (size_t)b * DIN + d0);
    float xs[4] = {x4.x, x4.y, x4.z, x4.w};
    float a0 = 0.f, a1 = 0.f, a2 = 0.f;
#pragma unroll
    for (int j = 0; j < 4; ++j) {
      int d = d0 + j;
      float xx = xs[j];
      float4 kv = kva[j];
      int si = (xx >= kv.x) + (xx >= kv.y) + (xx >= kv.z) + (xx >= kv.w) +
               (xx >= k8a[j]);
      const float4* Ap = (const float4*)(apoly + (size_t)(d * NSEG + si) * 12);
      float4 c0 = Ap[0], c1 = Ap[1], c2 = Ap[2];
      a0 += fmaf(fmaf(fmaf(c0.w, xx, c0.z), xx, c0.y), xx, c0.x);
      a1 += fmaf(fmaf(fmaf(c1.w, xx, c1.z), xx, c1.y), xx, c1.x);
      a2 += fmaf(fmaf(fmaf(c2.w, xx, c2.z), xx, c2.y), xx, c2.x);
    }
    // wave32 reduction over the 32 d-lanes
#pragma unroll
    for (int off = 16; off > 0; off >>= 1) {
      a0 += __shfl_down(a0, off, 32);
      a1 += __shfl_down(a1, off, 32);
      a2 += __shfl_down(a2, off, 32);
    }
    if (lane == 0) {
      float* p = part + ((size_t)dchunk * BATCH + b) * OOUT;
      p[0] = a0; p[1] = a1; p[2] = a2;
    }
  }
}

// ----------------------- out_base via WMMA (f16) ----------------------------
// out_base = silu(X) @ WbP : M=2048, K=4096, N=3 (padded to 16).
// block = 8 waves; each block owns one 16-row M tile; waves split K (512 each).
__global__ __launch_bounds__(256) void kan_base_wmma(
    const float* __restrict__ x, const float* __restrict__ wbp,
    float* __restrict__ part)
{
  __shared__ float red[8][16][16];
  int tid  = threadIdx.x;
  int lane = tid & 31;
  int w    = tid >> 5;
  int hi   = lane >> 4;                  // lane half (K split)
  int n    = lane & 15;                  // N column (also M row for A loads)
  int mbase = blockIdx.x * 16;

  const float* xr = x + (size_t)(mbase + n) * DIN;  // A row M = lane%16
  int nc = (n < 3) ? n : 3;              // pad column -> reads zero slot

  v8f acc = {};
#pragma unroll 4
  for (int it = 0; it < 16; ++it) {
    int kk = (w << 9) + (it << 5);       // this wave's K window of 32
    int K0 = kk + (hi << 3);             // A elems 0..7  -> K0..K0+7
    int K1 = kk + 16 + (hi << 3);        // A elems 8..15 -> K1..K1+7

    float4 r0 = *(const float4*)(xr + K0);
    float4 r1 = *(const float4*)(xr + K0 + 4);
    float4 r2 = *(const float4*)(xr + K1);
    float4 r3 = *(const float4*)(xr + K1 + 4);

    v16h a;
    float av[16] = {r0.x, r0.y, r0.z, r0.w, r1.x, r1.y, r1.z, r1.w,
                    r2.x, r2.y, r2.z, r2.w, r3.x, r3.y, r3.z, r3.w};
#pragma unroll
    for (int e = 0; e < 16; ++e) {
      float v = av[e];
      float sv = v * __builtin_amdgcn_rcpf(1.0f + __expf(-v));   // silu (native exp)
      a[e] = (_Float16)sv;
    }

    // B 32x16 f16: elems 0..7 hold K=8*hi+e, elems 8..15 hold K=16+8*hi+(e-8), col = n
    v16h bm;
#pragma unroll
    for (int e = 0; e < 16; ++e) {
      int K = kk + (hi << 3) + e + ((e >> 3) << 3);
      bm[e] = (_Float16)wbp[(size_t)K * 4 + nc];
    }

    acc = __builtin_amdgcn_wmma_f32_16x16x32_f16(
        /*neg_a=*/false, a, /*neg_b=*/false, bm,
        /*c_mod=*/(short)0, acc, /*reuse_a=*/false, /*reuse_b=*/false);
  }

  // combine the 8 waves' partial 16x16 tiles (D layout: M = i + 8*hi, N = n)
#pragma unroll
  for (int i = 0; i < 8; ++i) red[w][i + 8 * hi][n] = acc[i];
  __syncthreads();

  float s = 0.f;
#pragma unroll
  for (int ww = 0; ww < 8; ++ww) s += red[ww][tid >> 4][tid & 15];
  int m  = mbase + (tid >> 4);
  int nn = tid & 15;
  if (nn < 3)
    part[((size_t)32 * BATCH + m) * OOUT + nn] = s;
}

// ------------------------------ reduce --------------------------------------
__global__ __launch_bounds__(256) void kan_reduce(
    const float* __restrict__ part, float* __restrict__ out)
{
  int i = blockIdx.x * blockDim.x + threadIdx.x;
  if (i >= BATCH * OOUT) return;
  float s = 0.f;
#pragma unroll
  for (int c = 0; c < NPART; ++c) s += part[(size_t)c * (BATCH * OOUT) + i];
  out[i] = s;
}

// ------------------------------ launch --------------------------------------
extern "C" void kernel_launch(void* const* d_in, const int* in_sizes, int n_in,
                              void* d_out, int out_size, void* d_ws, size_t ws_size,
                              hipStream_t stream) {
  (void)in_sizes; (void)n_in; (void)out_size; (void)ws_size;
  const float* x          = (const float*)d_in[0];
  const float* grid       = (const float*)d_in[4];
  const float* coef       = (const float*)d_in[5];
  const float* scale_base = (const float*)d_in[6];
  const float* scale_sp   = (const float*)d_in[7];
  const float* mask       = (const float*)d_in[8];
  float* out = (float*)d_out;
  float* ws  = (float*)d_ws;

  kan_prep<<<(DIN * NSEG + 255) / 256, 256, 0, stream>>>(
      grid, coef, scale_base, scale_sp, mask, ws);
  kan_spline<<<256, 256, 0, stream>>>(x, ws, ws + PART_F);
  kan_base_wmma<<<BATCH / 16, 256, 0, stream>>>(x, ws + WBP_F, ws + PART_F);
  kan_reduce<<<(BATCH * OOUT + 255) / 256, 256, 0, stream>>>(ws + PART_F, out);
}
